// XorzoTransformer_59287728554100
// MI455X (gfx1250) — compile-verified, matrix operand order checked by reading, not run on hardware
//
#include <hip/hip_runtime.h>
#include <hip/hip_bf16.h>
#include <math.h>

// ---------------------------------------------------------------------------
// MI455X (gfx1250, wave32) transformer forward.
// All GEMMs run on v_wmma_f32_16x16x32_bf16 (bf16 in, fp32 accum).
// Weights stay fp32 in HBM (192MB L2 makes per-layer re-reads free) and are
// converted to bf16 while staging into LDS.
// ---------------------------------------------------------------------------

typedef __bf16 bf16;
typedef __attribute__((ext_vector_type(16))) __bf16 v16bf;
typedef __attribute__((ext_vector_type(8)))  float  v8f;
typedef __attribute__((ext_vector_type(4)))  float  f32x4;

#define D_    1024
#define T_    1024
#define B_    2
#define H_    16
#define DH_   64
#define L_    8
#define VOC_  32000
#define DFF_  1656
#define DFFP_ 1664
#define NTOK  (B_*T_)

union FragU { v16bf v; f32x4 f[2]; };

// Load one 16x(32) 16-bit A/B-style WMMA fragment from a [row][k] tile.
// ISA layout (16-bit A 16x32): lanes 0-15 hold k=0..7 (elems 0-7) and
// k=16..23 (elems 8-15); lanes 16-31 hold k=8..15 / k=24..31. Rows = lane&15.
__device__ __forceinline__ v16bf ld_frag(const bf16* tile, int lane, int stride) {
    const bf16* p = tile + (lane & 15) * stride + ((lane >> 4) << 3);
    FragU u;
    u.f[0] = *(const f32x4*)(p);
    u.f[1] = *(const f32x4*)(p + 16);
    return u.v;
}

__device__ __forceinline__ v8f wmma_bf16(v16bf a, v16bf b, v8f c) {
    return __builtin_amdgcn_wmma_f32_16x16x32_bf16(false, a, false, b, (short)0, c,
                                                   false, false);
}

__device__ __forceinline__ float gelu_exact(float x) {
    return 0.5f * x * (1.0f + erff(x * 0.70710678118654752f));
}

// ---------------------------------------------------------------------------
// Embedding + golden-angle positional encoding:  x[row] = embed[tok] + pe[t]
// ---------------------------------------------------------------------------
__global__ __launch_bounds__(256) void embed_kernel(
    const float* __restrict__ emb, const float* __restrict__ pe,
    const int* __restrict__ tok, float* __restrict__ x)
{
    int row = blockIdx.x;            // b*T + t
    int t   = row & (T_ - 1);
    int tk  = tok[row];
    int d   = threadIdx.x * 4;
    f32x4 e = *(const f32x4*)(emb + (size_t)tk * D_ + d);
    f32x4 p = *(const f32x4*)(pe + (size_t)t * D_ + d);
    *(f32x4*)(x + (size_t)row * D_ + d) = e + p;
}

// ---------------------------------------------------------------------------
// Balance-norm: layernorm + half-energy-imbalance correction, bf16 output.
// One block per token row, 256 threads x 4 elems (threads 0-127 = first half).
// ---------------------------------------------------------------------------
__global__ __launch_bounds__(256) void bnorm_kernel(
    const float* __restrict__ x, const float* __restrict__ g,
    const float* __restrict__ bv, const float* __restrict__ bwp,
    bf16* __restrict__ out)
{
    __shared__ float ss[256], qq[256];
    int row = blockIdx.x, tid = threadIdx.x;
    const float* xr = x + (size_t)row * D_;
    f32x4 v = *(const f32x4*)(xr + tid * 4);
    float s = v[0] + v[1] + v[2] + v[3];
    float q = v[0]*v[0] + v[1]*v[1] + v[2]*v[2] + v[3]*v[3];
    ss[tid] = s; qq[tid] = q;
    __syncthreads();
    // reduce each 128-thread half separately (qq halves give e1/e2)
    for (int st = 64; st > 0; st >>= 1) {
        if ((tid & 127) < st) {
            int i = (tid & 128) + (tid & 127);
            ss[i] += ss[i + st];
            qq[i] += qq[i + st];
        }
        __syncthreads();
    }
    float mean = (ss[0] + ss[128]) * (1.0f / D_);
    float e1   = qq[0]   * (2.0f / D_);
    float e2   = qq[128] * (2.0f / D_);
    float var  = (qq[0] + qq[128]) * (1.0f / D_) - mean * mean;
    float rstd = rsqrtf(var + 1e-6f);
    float imb  = (e1 - e2) * bwp[0];
    float corr = (tid < 128) ? -imb : imb;
    bf16* o = out + (size_t)row * D_ + tid * 4;
    #pragma unroll
    for (int j = 0; j < 4; ++j) {
        int d = tid * 4 + j;
        float y = (v[j] - mean) * rstd * g[d] + bv[d] + corr;
        o[j] = (bf16)y;
    }
}

// ---------------------------------------------------------------------------
// Generic WMMA GEMM:  C[M,N] = epi( A_bf16[M,K] * W_f32[K,N] + bias )
// Block tile 128x128, BK=32, 8 waves (2x4), wave tile 64x32 (4x2 WMMA tiles).
// W converted f32->bf16 while staging transposed [n][k] into LDS so both
// operands use contiguous ds_load_b128 fragment loads.
// ---------------------------------------------------------------------------
template<bool GELU, bool RESID, bool OUT_BF16>
__global__ __launch_bounds__(256) void gemm_kernel(
    const bf16*  __restrict__ A, int lda,
    const float* __restrict__ W,          // [K, N] row-major fp32
    const float* __restrict__ bias,       // nullable
    const float* __restrict__ resid,      // used iff RESID
    void* __restrict__ Cout, int ldc,
    int M, int N, int K)
{
    __shared__ alignas(16) bf16 As[128 * 48];   // [m][k], stride 48
    __shared__ alignas(16) bf16 Bs[128 * 48];   // [n][k], stride 48

    const int tid  = threadIdx.x;
    const int lane = tid & 31;
    const int wid  = tid >> 5;
    const int wm   = wid >> 2;      // 0..1  (64 rows each)
    const int wn   = wid & 3;       // 0..3  (32 cols each)
    const int m0   = blockIdx.y * 128;
    const int n0   = blockIdx.x * 128;

    v8f acc[4][2];
    #pragma unroll
    for (int i = 0; i < 4; ++i)
        #pragma unroll
        for (int j = 0; j < 2; ++j)
            #pragma unroll
            for (int e = 0; e < 8; ++e) acc[i][j][e] = 0.0f;

    for (int k0 = 0; k0 < K; k0 += 32) {
        // ---- stage A (bf16, vectorized 2x b128 per thread) ----
        {
            int r  = tid >> 1;
            int kg = (tid & 1) * 16;
            const bf16* src = A + (size_t)(m0 + r) * lda + k0 + kg;
            bf16* dst = &As[r * 48 + kg];
            if (k0 + 32 <= K) {
                *(f32x4*)(dst)     = *(const f32x4*)(src);
                *(f32x4*)(dst + 8) = *(const f32x4*)(src + 8);
            } else {
                #pragma unroll
                for (int j = 0; j < 16; ++j)
                    dst[j] = (k0 + kg + j < K) ? src[j] : (bf16)0.0f;
            }
        }
        // ---- stage W transposed, f32 -> bf16 ----
        {
            int kr = tid >> 3;             // 0..31
            int ng = (tid & 7) * 16;       // 0..112
            const float* src = W + (size_t)(k0 + kr) * N + n0 + ng;
            bool kin = (k0 + kr) < K;
            __builtin_prefetch(src + (size_t)32 * N, 0, 1);   // next K tile
            #pragma unroll
            for (int j = 0; j < 16; ++j) {
                int gn = n0 + ng + j;
                float w = (kin && gn < N) ? src[j] : 0.0f;
                Bs[(ng + j) * 48 + kr] = (bf16)w;
            }
        }
        __syncthreads();

        v16bf af[4], bfr[2];
        #pragma unroll
        for (int mi = 0; mi < 4; ++mi)
            af[mi] = ld_frag(&As[(wm * 64 + mi * 16) * 48], lane, 48);
        #pragma unroll
        for (int ni = 0; ni < 2; ++ni)
            bfr[ni] = ld_frag(&Bs[(wn * 32 + ni * 16) * 48], lane, 48);

        #pragma unroll
        for (int mi = 0; mi < 4; ++mi)
            #pragma unroll
            for (int ni = 0; ni < 2; ++ni)
                acc[mi][ni] = wmma_bf16(af[mi], bfr[ni], acc[mi][ni]);

        __syncthreads();
    }

    // ---- epilogue: C layout rows = vgpr idx (+8 for lanes 16-31), col = lane&15
    const int col16 = lane & 15;
    const int hi8   = (lane >> 4) << 3;
    #pragma unroll
    for (int mi = 0; mi < 4; ++mi) {
        #pragma unroll
        for (int ni = 0; ni < 2; ++ni) {
            #pragma unroll
            for (int v = 0; v < 8; ++v) {
                int row = m0 + wm * 64 + mi * 16 + v + hi8;
                int cn  = n0 + wn * 32 + ni * 16 + col16;
                if (row < M && cn < N) {
                    float x = acc[mi][ni][v];
                    if (bias)  x += bias[cn];
                    if (GELU)  x = gelu_exact(x);
                    if (RESID) x += resid[(size_t)row * ldc + cn];
                    if (OUT_BF16) ((bf16*)Cout)[(size_t)row * ldc + cn] = (bf16)x;
                    else          ((float*)Cout)[(size_t)row * ldc + cn] = x;
                }
            }
        }
    }
}

// ---------------------------------------------------------------------------
// Flash attention + golden-rotation epilogue.
// Grid (T/64, B*H); 128 threads = 4 waves, each wave owns 16 q rows x DH=64.
// Q and K^T fragments load straight from global (contiguous per lane);
// P goes through per-wave LDS for the C-layout -> A-layout transpose;
// V is staged transposed [dh][kpos] in LDS.
// ---------------------------------------------------------------------------
__global__ __launch_bounds__(128) void attn_kernel(
    const bf16* __restrict__ Q, const bf16* __restrict__ Kb,
    const bf16* __restrict__ Vb,
    const float* __restrict__ beta, const float* __restrict__ chi,
    bf16* __restrict__ Out)
{
    __shared__ alignas(16) bf16 Vs[64 * 80];        // [dh][kpos]
    __shared__ alignas(16) bf16 Ps[4][16 * 80];     // per-wave [qrow][kpos]

    const int tid  = threadIdx.x;
    const int lane = tid & 31;
    const int w    = tid >> 5;
    const int col  = lane & 15;
    const int hi   = lane >> 4;           // 0/1
    const int qt   = blockIdx.x;
    const int bh   = blockIdx.y;
    const int b    = bh >> 4, h = bh & 15;
    const size_t base = (size_t)b * T_ * D_ + h * DH_;
    const int q0 = qt * 64 + w * 16;

    // Q fragments (K-dim 64 -> two 16x32 chunks), direct from global
    v16bf qf[2];
    {
        const bf16* qp = Q + base + (size_t)q0 * D_;
        #pragma unroll
        for (int c = 0; c < 2; ++c)
            qf[c] = ld_frag(qp + c * 32, lane, D_);
    }

    float mrun[8], lrun[8];
    v8f cacc[4];
    #pragma unroll
    for (int v = 0; v < 8; ++v) { mrun[v] = -1e30f; lrun[v] = 0.0f; }
    #pragma unroll
    for (int ni = 0; ni < 4; ++ni)
        #pragma unroll
        for (int e = 0; e < 8; ++e) cacc[ni][e] = 0.0f;

    const int ktend = qt * 64 + 64;       // causal upper bound for this block
    for (int kt = 0; kt < ktend; kt += 64) {
        // stage V tile transposed: [dh][kpos]
        {
            int kp  = tid >> 1;
            int dh0 = (tid & 1) * 32;
            const bf16* vp = Vb + base + (size_t)(kt + kp) * D_ + dh0;
            #pragma unroll
            for (int j = 0; j < 32; ++j)
                Vs[(dh0 + j) * 80 + kp] = vp[j];
        }
        __syncthreads();

        // S = Q K^T : 4 col-tiles x (K=64 via 2 chained wmma)
        v8f sacc[4];
        #pragma unroll
        for (int s = 0; s < 4; ++s)
            #pragma unroll
            for (int e = 0; e < 8; ++e) sacc[s][e] = 0.0f;
        #pragma unroll
        for (int sub = 0; sub < 4; ++sub) {
            const bf16* kp = Kb + base + (size_t)(kt + sub * 16) * D_;
            #pragma unroll
            for (int c = 0; c < 2; ++c) {
                v16bf kf = ld_frag(kp + c * 32, lane, D_);
                sacc[sub] = wmma_bf16(qf[c], kf, sacc[sub]);
            }
        }

        // online softmax (rows: vgpr idx + 8*hi; cols: 16-lane groups)
        #pragma unroll
        for (int v = 0; v < 8; ++v) {
            int qrow = q0 + v + hi * 8;
            float mx = -1e30f;
            #pragma unroll
            for (int sub = 0; sub < 4; ++sub) {
                int kc = kt + sub * 16 + col;
                float s = sacc[sub][v] * 0.125f;           // 1/sqrt(DH)
                if (kc > qrow) s = -1e30f;                 // causal mask
                sacc[sub][v] = s;
                mx = fmaxf(mx, s);
            }
            #pragma unroll
            for (int off = 8; off > 0; off >>= 1)
                mx = fmaxf(mx, __shfl_xor(mx, off, 16));
            float mnew  = fmaxf(mrun[v], mx);
            float alpha = __expf(mrun[v] - mnew);
            #pragma unroll
            for (int ni = 0; ni < 4; ++ni) cacc[ni][v] *= alpha;
            float psum = 0.0f;
            #pragma unroll
            for (int sub = 0; sub < 4; ++sub) {
                float p = __expf(sacc[sub][v] - mnew);
                psum += p;
                Ps[w][(v + hi * 8) * 80 + sub * 16 + col] = (bf16)p;
            }
            #pragma unroll
            for (int off = 8; off > 0; off >>= 1)
                psum += __shfl_xor(psum, off, 16);
            lrun[v] = lrun[v] * alpha + psum;
            mrun[v] = mnew;
        }

        asm volatile("s_wait_dscnt 0" ::: "memory");   // P: in-wave DS RAW

        // ctx += P V  (K-dim 64 -> 2 chunks; 4 dh tiles)
        v16bf pf[2];
        #pragma unroll
        for (int c = 0; c < 2; ++c)
            pf[c] = ld_frag(&Ps[w][c * 32], lane, 80);
        #pragma unroll
        for (int ni = 0; ni < 4; ++ni) {
            #pragma unroll
            for (int c = 0; c < 2; ++c) {
                v16bf vf = ld_frag(&Vs[(ni * 16) * 80 + c * 32], lane, 80);
                cacc[ni] = wmma_bf16(pf[c], vf, cacc[ni]);
            }
        }
        __syncthreads();
    }

    // epilogue: normalize, golden rotation, tanh(chi) gate, bf16 store
    float sg  = 1.0f / (1.0f + __expf(-beta[h]));
    float ang = 3.14159265358979323846f * sg;
    float cc  = __cosf(ang), sn = __sinf(ang);
    float ch  = tanhf(chi[h]);
    #pragma unroll
    for (int p0 = 0; p0 < 2; ++p0) {        // dh tiles (p0, p0+2) pair re/im
        int p1 = p0 + 2;
        #pragma unroll
        for (int v = 0; v < 8; ++v) {
            float inv = 1.0f / lrun[v];
            float xr = cacc[p0][v] * inv;
            float xi = cacc[p1][v] * inv;
            float rr = (xr * cc - xi * sn) * ch;
            float ri = (xr * sn + xi * cc) * ch;
            int trow = q0 + v + hi * 8;
            bf16* o = Out + base + (size_t)trow * D_;
            o[p0 * 16 + col] = (bf16)rr;
            o[p1 * 16 + col] = (bf16)ri;
        }
    }
}

// ---------------------------------------------------------------------------
// Launch
// ---------------------------------------------------------------------------
extern "C" void kernel_launch(void* const* d_in, const int* in_sizes, int n_in,
                              void* d_out, int out_size, void* d_ws, size_t ws_size,
                              hipStream_t stream)
{
    (void)in_sizes; (void)n_in; (void)out_size; (void)ws_size;
    const float* emb = (const float*)d_in[0];
    const float* pe  = (const float*)d_in[1];
    auto P = [&](int l, int j) { return d_in[2 + l * 16 + j]; };
    const int tail = 2 + L_ * 16;
    const float* nf_g   = (const float*)d_in[tail + 0];
    const float* nf_b   = (const float*)d_in[tail + 1];
    const float* nf_bw  = (const float*)d_in[tail + 2];
    const float* w_head = (const float*)d_in[tail + 3];
    const float* b_head = (const float*)d_in[tail + 4];
    const int*   tokens = (const int*)  d_in[tail + 5];

    char* w = (char*)d_ws;
    float* x  = (float*)w; w += (size_t)NTOK * D_ * 4;
    bf16* xn  = (bf16*)w;  w += (size_t)NTOK * D_ * 2;
    bf16* qb  = (bf16*)w;  w += (size_t)NTOK * D_ * 2;
    bf16* kb  = (bf16*)w;  w += (size_t)NTOK * D_ * 2;
    bf16* vb  = (bf16*)w;  w += (size_t)NTOK * D_ * 2;
    bf16* ab  = (bf16*)w;  w += (size_t)NTOK * D_ * 2;
    bf16* hb  = (bf16*)w;  w += (size_t)NTOK * DFFP_ * 2;

    const dim3 blk256(256), blk128(128);
    const dim3 gD(D_ / 128, NTOK / 128);        // 8 x 16
    const dim3 gFF((DFF_ + 127) / 128, NTOK / 128);
    const dim3 gHead(VOC_ / 128, NTOK / 128);   // 250 x 16
    const dim3 gAttn(T_ / 64, B_ * H_);

    embed_kernel<<<NTOK, blk256, 0, stream>>>(emb, pe, tokens, x);

    for (int l = 0; l < L_; ++l) {
        const float* wq = (const float*)P(l, 3);
        const float* wk = (const float*)P(l, 4);
        const float* wv = (const float*)P(l, 5);
        const float* wo = (const float*)P(l, 6);

        bnorm_kernel<<<NTOK, blk256, 0, stream>>>(
            x, (const float*)P(l, 0), (const float*)P(l, 1),
            (const float*)P(l, 2), xn);

        gemm_kernel<false, false, true><<<gD, blk256, 0, stream>>>(
            xn, D_, wq, nullptr, nullptr, qb, D_, NTOK, D_, D_);
        gemm_kernel<false, false, true><<<gD, blk256, 0, stream>>>(
            xn, D_, wk, nullptr, nullptr, kb, D_, NTOK, D_, D_);
        gemm_kernel<false, false, true><<<gD, blk256, 0, stream>>>(
            xn, D_, wv, nullptr, nullptr, vb, D_, NTOK, D_, D_);

        attn_kernel<<<gAttn, blk128, 0, stream>>>(
            qb, kb, vb, (const float*)P(l, 7), (const float*)P(l, 8), ab);

        // x += attn @ wo   (residual fused)
        gemm_kernel<false, true, false><<<gD, blk256, 0, stream>>>(
            ab, D_, wo, nullptr, x, x, D_, NTOK, D_, D_);

        bnorm_kernel<<<NTOK, blk256, 0, stream>>>(
            x, (const float*)P(l, 9), (const float*)P(l, 10),
            (const float*)P(l, 11), xn);

        // h = gelu(xn @ w_in + b_in)  -> bf16 [NTOK, DFFP]
        gemm_kernel<true, false, true><<<gFF, blk256, 0, stream>>>(
            xn, D_, (const float*)P(l, 12), (const float*)P(l, 13), nullptr,
            hb, DFFP_, NTOK, DFF_, D_);

        // x += h @ w_out + b_out
        gemm_kernel<false, true, false><<<gD, blk256, 0, stream>>>(
            hb, DFFP_, (const float*)P(l, 14), (const float*)P(l, 15), x,
            x, D_, NTOK, D_, DFF_);
    }

    bnorm_kernel<<<NTOK, blk256, 0, stream>>>(x, nf_g, nf_b, nf_bw, xn);

    // logits = xn @ w_head + b_head  (fp32 out)
    gemm_kernel<false, false, false><<<gHead, blk256, 0, stream>>>(
        xn, D_, w_head, b_head, nullptr, d_out, VOC_, NTOK, VOC_, D_);
}
